// HingebasedCrossAttentionCLIP_88132728914462
// MI455X (gfx1250) — matrix operand order repaired and checked by reading, not compiled, not run on hardware
//
#include <hip/hip_runtime.h>
#include <hip/hip_bf16.h>
#include <stdint.h>

// ---------------------------------------------------------------------------
// Problem constants
// ---------------------------------------------------------------------------
#define BATCH 256
#define DIM   640
#define LRr   257   // reference tokens (only row 0 used)
#define LCc   77    // caption tokens
#define LTt   257   // target tokens
#define SCALE_F (0.03952847075210474f)  // 1/sqrt(640)

typedef __attribute__((ext_vector_type(16))) __bf16 v16bf;
typedef __attribute__((ext_vector_type(8)))  float  v8f;

__device__ __forceinline__ uint16_t f2bf(float f) {
    uint32_t u = __float_as_uint(f);
    uint32_t r = u + 0x7FFFu + ((u >> 16) & 1u);   // round-to-nearest-even
    return (uint16_t)(r >> 16);
}

// ---------------------------------------------------------------------------
// Prep kernels: f32 -> bf16 conversions / transpose / row-0 gather
// ---------------------------------------------------------------------------
__global__ void k_f32_to_bf16(const float* __restrict__ src,
                              uint16_t* __restrict__ dst, int n) {
    int i = blockIdx.x * 256 + threadIdx.x;
    if (i < n) dst[i] = f2bf(src[i]);
}

// dst[r*DIM + c] = bf16(src[c*DIM + r])   (640x640 transpose)
__global__ void k_f32_to_bf16_T(const float* __restrict__ src,
                                uint16_t* __restrict__ dst) {
    int i = blockIdx.x * 256 + threadIdx.x;
    if (i < DIM * DIM) {
        int r = i / DIM, c = i % DIM;
        dst[i] = f2bf(src[c * DIM + r]);
    }
}

// R0h[b,d] = bf16(ref[b, 0, d])
__global__ void k_gather_r0(const float* __restrict__ ref,
                            uint16_t* __restrict__ dst) {
    int i = blockIdx.x * 256 + threadIdx.x;   // i < BATCH*DIM
    int b = i / DIM, d = i % DIM;
    dst[i] = f2bf(ref[(size_t)b * LRr * DIM + d]);
}

// ---------------------------------------------------------------------------
// WMMA GEMM:  C[M,N] = A[M,K](bf16) * Bt[N,K](bf16, pre-transposed B)
//             + biasScale[m] * bias[n]
// One wave per 16x16 tile; 4 waves (N-direction) per 128-thread block.
// ---------------------------------------------------------------------------
union AFrag { v16bf v; uint4 q[2]; };

__global__ void k_gemm_bf16_wmma(const uint16_t* __restrict__ A,
                                 const uint16_t* __restrict__ Bt,
                                 float*    __restrict__ Cf,     // may be null
                                 uint16_t* __restrict__ Ch,     // may be null
                                 const float* __restrict__ bias,      // [N] or null
                                 const float* __restrict__ biasScale, // [M] or null
                                 int M, int N, int K) {
    const int lane = threadIdx.x & 31;
    const int wave = threadIdx.x >> 5;
    const int tn   = blockIdx.x * 4 + wave;     // N tile
    const int tm   = blockIdx.y;                // M tile
    const int hf   = lane >> 4;                 // lane half (0/1)
    const int l15  = lane & 15;
    const int m    = tm * 16 + l15;
    const int n    = tn * 16 + l15;

    const uint16_t* arow = A  + (size_t)m * K;
    const uint16_t* brow = Bt + (size_t)n * K;

    v8f acc = {0.f, 0.f, 0.f, 0.f, 0.f, 0.f, 0.f, 0.f};
    for (int k = 0; k < K; k += 32) {
        AFrag a, b;
        // A 16x32 bf16 layout: half 0 -> K 0-7 & 16-23, half 1 -> K 8-15 & 24-31
        a.q[0] = *(const uint4*)(arow + k + hf * 8);
        a.q[1] = *(const uint4*)(arow + k + 16 + hf * 8);
        // B 32x16 bf16 layout: lane holds column n, half 0 -> K 0-7 & 16-23, half 1 -> K 8-15 & 24-31
        b.q[0] = *(const uint4*)(brow + k + hf * 8);
        b.q[1] = *(const uint4*)(brow + k + 16 + hf * 8);
        acc = __builtin_amdgcn_wmma_f32_16x16x32_bf16(
                  false, a.v, false, b.v, (short)0, acc, false, false);
    }

#pragma unroll
    for (int r = 0; r < 8; ++r) {
        int mo = tm * 16 + hf * 8 + r;          // C layout: M = 8*half + r
        float v = acc[r];
        if (bias) {
            float bs = biasScale ? biasScale[mo] : 1.0f;
            v += bs * bias[n];
        }
        size_t idx = (size_t)mo * N + n;
        if (Cf) Cf[idx] = v;
        if (Ch) Ch[idx] = f2bf(v);
    }
}

// ---------------------------------------------------------------------------
// K3: per-batch caption pass.
//   qb      = Q0[b] . bkt
//   attA[c] = SCALE * (P[b] . cap[b,c] + qb)            c < 77
//   sumA[b] = sum_c attA[c]
//   S_ch[b] = bf16( sum_c attA[c] * cap[b,c,:] )
// ---------------------------------------------------------------------------
__global__ void k_caption_pass(const float* __restrict__ cap,   // [B,LC,D]
                               const float* __restrict__ Pf,    // [B,D]
                               const float* __restrict__ Q0f,   // [B,D]
                               const float* __restrict__ bkt,   // [D]
                               uint16_t* __restrict__ S_ch,     // [B,D] bf16
                               float* __restrict__ sumA) {      // [B]
    const int b = blockIdx.x, tid = threadIdx.x;
    const int lane = tid & 31, wave = tid >> 5;
    __shared__ float sP[DIM];
    __shared__ float sA[LCc];
    __shared__ float red[256];
    const float* capb = cap + (size_t)b * LCc * DIM;

    float qb_part = 0.f;
    for (int d = tid; d < DIM; d += 256) {
        sP[d] = Pf[(size_t)b * DIM + d];
        qb_part += Q0f[(size_t)b * DIM + d] * bkt[d];
    }
    red[tid] = qb_part; __syncthreads();
    for (int s = 128; s > 0; s >>= 1) { if (tid < s) red[tid] += red[tid + s]; __syncthreads(); }
    const float qb = red[0];
    __syncthreads();

    for (int c = wave; c < LCc; c += 8) {
        const float* row = capb + c * DIM;
        __builtin_prefetch(capb + (c + 8) * DIM + lane * 4, 0, 1);
        float acc = 0.f;
        for (int d = lane; d < DIM; d += 32) acc += sP[d] * row[d];
        for (int off = 16; off > 0; off >>= 1) acc += __shfl_down(acc, off);
        if (lane == 0) sA[c] = SCALE_F * (acc + qb);
    }
    __syncthreads();

    float sa = 0.f;
    for (int c = tid; c < LCc; c += 256) sa += sA[c];
    red[tid] = sa; __syncthreads();
    for (int s = 128; s > 0; s >>= 1) { if (tid < s) red[tid] += red[tid + s]; __syncthreads(); }
    if (tid == 0) sumA[b] = red[0];

    for (int d = tid; d < DIM; d += 256) {
        float acc = 0.f;
        for (int c = 0; c < LCc; ++c) acc += sA[c] * capb[c * DIM + d];
        S_ch[(size_t)b * DIM + d] = f2bf(acc);
    }
}

// ---------------------------------------------------------------------------
// K5: per-batch target pass.
//   cb        = SCALE * (Wmat[b] . bktar)
//   scores[t] = SCALE * (U[b] . tar[b,t]) + cb
//   attn      = softmax(scores)
//   Yh[b]     = bf16( sum_t attn[t] * tar[b,t,:] )
// ---------------------------------------------------------------------------
__global__ void k_target_pass(const float* __restrict__ tar,    // [B,LT,D]
                              const float* __restrict__ Uf,     // [B,D]
                              const float* __restrict__ Wmatf,  // [B,D]
                              const float* __restrict__ bktar,  // [D]
                              uint16_t* __restrict__ Yh) {      // [B,D] bf16
    const int b = blockIdx.x, tid = threadIdx.x;
    const int lane = tid & 31, wave = tid >> 5;
    __shared__ float sU[DIM];
    __shared__ float sS[LTt];
    __shared__ float red[256];
    const float* tb = tar + (size_t)b * LTt * DIM;

    float cb_part = 0.f;
    for (int d = tid; d < DIM; d += 256) {
        sU[d] = Uf[(size_t)b * DIM + d];
        cb_part += Wmatf[(size_t)b * DIM + d] * bktar[d];
    }
    red[tid] = cb_part; __syncthreads();
    for (int s = 128; s > 0; s >>= 1) { if (tid < s) red[tid] += red[tid + s]; __syncthreads(); }
    const float cb = SCALE_F * red[0];
    __syncthreads();

    for (int t = wave; t < LTt; t += 8) {
        const float* row = tb + t * DIM;
        __builtin_prefetch(tb + (t + 8) * DIM + lane * 4, 0, 1);
        float acc = 0.f;
        for (int d = lane; d < DIM; d += 32) acc += sU[d] * row[d];
        for (int off = 16; off > 0; off >>= 1) acc += __shfl_down(acc, off);
        if (lane == 0) sS[t] = SCALE_F * acc + cb;
    }
    __syncthreads();

    // softmax over 257
    float m = -3.4e38f;
    for (int t = tid; t < LTt; t += 256) m = fmaxf(m, sS[t]);
    red[tid] = m; __syncthreads();
    for (int s = 128; s > 0; s >>= 1) { if (tid < s) red[tid] = fmaxf(red[tid], red[tid + s]); __syncthreads(); }
    const float mx = red[0];
    __syncthreads();

    float sum = 0.f;
    for (int t = tid; t < LTt; t += 256) { float e = __expf(sS[t] - mx); sS[t] = e; sum += e; }
    red[tid] = sum; __syncthreads();
    for (int s = 128; s > 0; s >>= 1) { if (tid < s) red[tid] += red[tid + s]; __syncthreads(); }
    const float inv = 1.0f / red[0];
    __syncthreads();
    for (int t = tid; t < LTt; t += 256) sS[t] *= inv;
    __syncthreads();

    for (int d = tid; d < DIM; d += 256) {
        float acc = 0.f;
        for (int t = 0; t < LTt; ++t) acc += sS[t] * tb[t * DIM + d];
        Yh[(size_t)b * DIM + d] = f2bf(acc);
    }
}

// ---------------------------------------------------------------------------
// Host-side orchestration
// ---------------------------------------------------------------------------
static inline size_t align256(size_t x) { return (x + 255) & ~(size_t)255; }

extern "C" void kernel_launch(void* const* d_in, const int* in_sizes, int n_in,
                              void* d_out, int out_size, void* d_ws, size_t ws_size,
                              hipStream_t stream) {
    (void)in_sizes; (void)n_in; (void)out_size; (void)ws_size;

    const float* ref   = (const float*)d_in[0];   // [B,257,640]
    const float* cap   = (const float*)d_in[1];   // [B,77,640]
    const float* tar   = (const float*)d_in[2];   // [B,257,640]
    const float* Wq    = (const float*)d_in[3];   // [640,640]
    const float* bq    = (const float*)d_in[4];
    const float* Wkt   = (const float*)d_in[5];
    const float* bkt   = (const float*)d_in[6];
    const float* Wktar = (const float*)d_in[7];
    const float* bktar = (const float*)d_in[8];
    const float* Wv    = (const float*)d_in[9];
    const float* bv    = (const float*)d_in[10];
    float* out = (float*)d_out;                   // [B,640]

    // workspace carve-up
    char* p = (char*)d_ws;
    const size_t szW  = (size_t)DIM * DIM * sizeof(uint16_t);   // 819200
    const size_t szBh = (size_t)BATCH * DIM * sizeof(uint16_t); // 327680
    const size_t szBf = (size_t)BATCH * DIM * sizeof(float);    // 655360
    uint16_t* WqH     = (uint16_t*)p; p += align256(szW);
    uint16_t* WktH    = (uint16_t*)p; p += align256(szW);
    uint16_t* WktTH   = (uint16_t*)p; p += align256(szW);   // transpose(Wkt)
    uint16_t* WktarTH = (uint16_t*)p; p += align256(szW);   // transpose(Wktar)
    uint16_t* WvH     = (uint16_t*)p; p += align256(szW);
    uint16_t* R0h     = (uint16_t*)p; p += align256(szBh);
    float*    Q0f     = (float*)p;    p += align256(szBf);
    uint16_t* Q0h     = (uint16_t*)p; p += align256(szBh);
    float*    Pf      = (float*)p;    p += align256(szBf);
    uint16_t* S_ch    = (uint16_t*)p; p += align256(szBh);
    float*    sumA    = (float*)p;    p += align256((size_t)BATCH * sizeof(float));
    float*    Wmatf   = (float*)p;    p += align256(szBf);
    uint16_t* Wmath   = (uint16_t*)p; p += align256(szBh);
    float*    Uf      = (float*)p;    p += align256(szBf);
    uint16_t* Yh      = (uint16_t*)p; p += align256(szBh);

    const int nW = DIM * DIM;                    // 409600
    dim3 cg((nW + 255) / 256), cb(256);
    // prep: bf16 weights (+ the two true transposes)
    k_f32_to_bf16   <<<cg, cb, 0, stream>>>(Wq,  WqH,  nW);
    k_f32_to_bf16   <<<cg, cb, 0, stream>>>(Wkt, WktH, nW);
    k_f32_to_bf16   <<<cg, cb, 0, stream>>>(Wv,  WvH,  nW);
    k_f32_to_bf16_T <<<cg, cb, 0, stream>>>(Wkt,   WktTH);
    k_f32_to_bf16_T <<<cg, cb, 0, stream>>>(Wktar, WktarTH);
    k_gather_r0     <<<dim3((BATCH * DIM) / 256), cb, 0, stream>>>(ref, R0h);

    dim3 ggrid(DIM / 64, BATCH / 16), gblk(128);  // 4 waves/block, 16x16 tile/wave
    // G1: Q0 = R0 @ Wq^T + bq            (Bt = Wq as stored)
    k_gemm_bf16_wmma<<<ggrid, gblk, 0, stream>>>(R0h, WqH, Q0f, Q0h, bq, nullptr,
                                                 BATCH, DIM, DIM);
    // G2: P = Q0 @ Wkt                   (Bt = Wkt^T)
    k_gemm_bf16_wmma<<<ggrid, gblk, 0, stream>>>(Q0h, WktTH, Pf, nullptr, nullptr, nullptr,
                                                 BATCH, DIM, DIM);
    // K3: attA / sumA / S_c over captions
    k_caption_pass<<<dim3(BATCH), dim3(256), 0, stream>>>(cap, Pf, Q0f, bkt, S_ch, sumA);
    // G4: Wmat = S_c @ Wkt^T + sumA*bkt  (Bt = Wkt as stored)
    k_gemm_bf16_wmma<<<ggrid, gblk, 0, stream>>>(S_ch, WktH, Wmatf, Wmath, bkt, sumA,
                                                 BATCH, DIM, DIM);
    // G5: U = Wmat @ Wktar               (Bt = Wktar^T)
    k_gemm_bf16_wmma<<<ggrid, gblk, 0, stream>>>(Wmath, WktarTH, Uf, nullptr, nullptr, nullptr,
                                                 BATCH, DIM, DIM);
    // K5: scores / softmax / Y over targets
    k_target_pass<<<dim3(BATCH), dim3(256), 0, stream>>>(tar, Uf, Wmatf, bktar, Yh);
    // G7: out = Y @ Wv^T + bv            (Bt = Wv as stored)
    k_gemm_bf16_wmma<<<ggrid, gblk, 0, stream>>>(Yh, WvH, out, nullptr, bv, nullptr,
                                                 BATCH, DIM, DIM);
}
